// SpatialDistanceGraph_31662498906306
// MI455X (gfx1250) — compile-verified, matrix operand order checked by reading, not run on hardware
//
#include <hip/hip_runtime.h>
#include <hip/hip_bf16.h>

// ---------------------------------------------------------------------------
// GAT forward for MI455X (gfx1250, wave32).
// Heavy GEMMs (57344x768x768, x5) on v_wmma_f32_16x16x32_bf16, fp32 accum.
// Inter-layer activations kept in bf16 in HBM (they are bf16-rounded for the
// WMMA anyway) -> halves the dominant memory traffic. Wh stays fp32 for the
// softmax path. Tile staging is software-pipelined through registers so next
// tile's global loads overlap the current tile's WMMAs.
// ---------------------------------------------------------------------------

typedef __attribute__((ext_vector_type(16))) __bf16 v16bf;
typedef __attribute__((ext_vector_type(8)))  float  v8f;

#define KDIM 768
#define NDIM 768
#define LDST 40          // LDS row stride in ushorts (80B: 16B-aligned, staggered banks)

__device__ __forceinline__ unsigned short f32_to_bf16_rne(float f) {
    unsigned int u = __float_as_uint(f);
    unsigned int r = (u + 0x7FFFu + ((u >> 16) & 1u)) >> 16;
    return (unsigned short)r;
}

union FragU { v16bf v; uint4 q[2]; };
union U16x16 { uint4 q[2]; unsigned short s[16]; };

// ---------------------------------------------------------------------------
// elementwise fp32 -> bf16 (weights, tiny)
// ---------------------------------------------------------------------------
__global__ void f32_to_bf16_kernel(const float* __restrict__ src,
                                   unsigned short* __restrict__ dst, int n) {
    int i = blockIdx.x * 256 + threadIdx.x;
    if (i < n) dst[i] = f32_to_bf16_rne(src[i]);
}

// ---------------------------------------------------------------------------
// C[M x 768] = A[M x 768] * Bw[768 x 768](bf16) (+ bias)
//   ABF16:   A is bf16 (ushort) else fp32 (converted while staging)
//   OUTBF16: C is bf16 (ushort) else fp32
// Block: 256 threads (8 waves), 128x128 tile. Wave: 32x64 -> 2x4 WMMA accums.
// ---------------------------------------------------------------------------
template<bool ABF16, bool OUTBF16>
__global__ __launch_bounds__(256) void gemm_wmma_768(
    const void* __restrict__ Av, const unsigned short* __restrict__ Bw,
    const float* __restrict__ bias, void* __restrict__ Cv, int useBias)
{
    __shared__ unsigned short ldsA[128 * LDST];     // [m][k]
    __shared__ unsigned short ldsB[128 * LDST];     // [n][k]  (B^T, matches A frag layout)

    const int tid  = threadIdx.x;
    const int lane = tid & 31;
    const int w    = tid >> 5;
    const int wm   = w & 3;          // 4 waves along M
    const int wn   = w >> 2;         // 2 waves along N
    const int m0   = blockIdx.y * 128;
    const int n0   = blockIdx.x * 128;
    const int r    = lane & 15;
    const int half = lane >> 4;      // lanes 16-31: +8 K-group / +8 M rows

    v8f zero = {};
    v8f acc[2][4];
#pragma unroll
    for (int mi = 0; mi < 2; ++mi)
#pragma unroll
        for (int ni = 0; ni < 4; ++ni) acc[mi][ni] = zero;

    const int arow = tid >> 1;          // 0..127
    const int akk  = (tid & 1) * 16;    // 0 / 16
    const int bkk  = tid >> 3;          // 0..31
    const int bns  = (tid & 7) * 16;    // 0..112

    U16x16 au, bu;   // register-staged tiles (16 bf16 each)

    // --- tile loaders (global -> regs) ---
    auto loadA = [&](int k0) {
        if (ABF16) {
            const unsigned short* ap =
                (const unsigned short*)Av + (size_t)(m0 + arow) * KDIM + k0 + akk;
            au.q[0] = ((const uint4*)ap)[0];
            au.q[1] = ((const uint4*)ap)[1];
        } else {
            const float* ap = (const float*)Av + (size_t)(m0 + arow) * KDIM + k0 + akk;
#pragma unroll
            for (int v4 = 0; v4 < 4; ++v4) {
                float4 f = ((const float4*)ap)[v4];
                au.s[v4 * 4 + 0] = f32_to_bf16_rne(f.x);
                au.s[v4 * 4 + 1] = f32_to_bf16_rne(f.y);
                au.s[v4 * 4 + 2] = f32_to_bf16_rne(f.z);
                au.s[v4 * 4 + 3] = f32_to_bf16_rne(f.w);
            }
        }
    };
    auto loadB = [&](int k0) {
        const unsigned short* bp = Bw + (size_t)(k0 + bkk) * NDIM + n0 + bns;
        bu.q[0] = ((const uint4*)bp)[0];
        bu.q[1] = ((const uint4*)bp)[1];
    };

    loadA(0);
    loadB(0);

    for (int k0 = 0; k0 < KDIM; k0 += 32) {
        // --- regs -> LDS ---
        {
            uint4* da = (uint4*)&ldsA[arow * LDST + akk];
            da[0] = au.q[0];
            da[1] = au.q[1];
#pragma unroll
            for (int j = 0; j < 16; ++j) ldsB[(bns + j) * LDST + bkk] = bu.s[j];  // transpose
        }
        __syncthreads();

        // --- issue next tile's global loads (overlap with WMMA below) ---
        if (k0 + 32 < KDIM) { loadA(k0 + 32); loadB(k0 + 32); }

        // --- fragments: two b128 LDS loads per frag per lane ---
        // 16-bit A 16x32 layout: lanes 0-15 -> K 0..7 (v0-3), 16..23 (v4-7);
        //                        lanes 16-31 -> K 8..15, 24..31.
        FragU afr[2], bfr[4];
#pragma unroll
        for (int mi = 0; mi < 2; ++mi) {
            const unsigned short* p = &ldsA[(wm * 32 + mi * 16 + r) * LDST + half * 8];
            afr[mi].q[0] = *(const uint4*)p;
            afr[mi].q[1] = *(const uint4*)(p + 16);
        }
#pragma unroll
        for (int ni = 0; ni < 4; ++ni) {
            const unsigned short* p = &ldsB[(wn * 64 + ni * 16 + r) * LDST + half * 8];
            bfr[ni].q[0] = *(const uint4*)p;
            bfr[ni].q[1] = *(const uint4*)(p + 16);
        }

#pragma unroll
        for (int mi = 0; mi < 2; ++mi)
#pragma unroll
            for (int ni = 0; ni < 4; ++ni)
                acc[mi][ni] = __builtin_amdgcn_wmma_f32_16x16x32_bf16(
                    false, afr[mi].v, false, bfr[ni].v,
                    (short)0, acc[mi][ni], false, false);

        __syncthreads();
    }

    // --- epilogue: C/D layout -> lanes 0-15: M=v, lanes 16-31: M=v+8; N=lane&15 ---
#pragma unroll
    for (int mi = 0; mi < 2; ++mi) {
#pragma unroll
        for (int ni = 0; ni < 4; ++ni) {
            const int col = n0 + wn * 64 + ni * 16 + r;
            const float bv = useBias ? bias[col] : 0.0f;
            const int rowBase = m0 + wm * 32 + mi * 16 + half * 8;
#pragma unroll
            for (int v = 0; v < 8; ++v) {
                const float val = acc[mi][ni][v] + bv;
                const size_t idx = (size_t)(rowBase + v) * NDIM + col;
                if (OUTBF16) ((unsigned short*)Cv)[idx] = f32_to_bf16_rne(val);
                else         ((float*)Cv)[idx] = val;
            }
        }
    }
}

// ---------------------------------------------------------------------------
// Fused GAT attention for one graph (block per graph, 256 threads):
//   ei/ej wave reductions, masked softmax over 14 neighbors,
//   h = attn @ Wh, optional ELU, attn -> output tuple, h -> bf16 activations.
// ---------------------------------------------------------------------------
__global__ __launch_bounds__(256) void gat_attention(
    const float* __restrict__ Wh, const float* __restrict__ adj,
    const float* __restrict__ avec,    // gat_a[l]: a_i = avec[0..767], a_j = avec[768..1535]
    unsigned short* __restrict__ x_out, float* __restrict__ attn_out, int apply_elu)
{
    __shared__ float sWh[14 * 768];    // 43 KB
    __shared__ float sAdj[196];
    __shared__ float sEi[14];
    __shared__ float sEj[14];
    __shared__ float sAttn[196];

    const int b   = blockIdx.x;
    const int tid = threadIdx.x;
    const size_t base = (size_t)b * (14 * 768);

    for (int idx = tid; idx < 14 * 768; idx += 256) sWh[idx] = Wh[base + idx];
    for (int idx = tid; idx < 196; idx += 256)      sAdj[idx] = adj[(size_t)b * 196 + idx];
    __syncthreads();

    // ei[i] = Wh[i,:].a_i ; ej[i] = Wh[i,:].a_j  (one wave per row, wave32 reduce)
    const int w = tid >> 5, lane = tid & 31;
    for (int i = w; i < 14; i += 8) {
        float pi = 0.0f, pj = 0.0f;
        for (int d = lane; d < 768; d += 32) {
            float x = sWh[i * 768 + d];
            pi += x * avec[d];
            pj += x * avec[768 + d];
        }
#pragma unroll
        for (int off = 16; off > 0; off >>= 1) {
            pi += __shfl_down(pi, off, 32);
            pj += __shfl_down(pj, off, 32);
        }
        if (lane == 0) { sEi[i] = pi; sEj[i] = pj; }
    }
    __syncthreads();

    // masked softmax over j (adj always has self-loop -> row never empty)
    if (tid < 14) {
        const int i = tid;
        float e[14];
        float mx = -1e30f;
#pragma unroll
        for (int j = 0; j < 14; ++j) {
            float v = (sAdj[i * 14 + j] > 0.0f) ? (sEi[i] + sEj[j]) : -1e30f;
            e[j] = v;
            mx = fmaxf(mx, v);
        }
        float s = 0.0f;
#pragma unroll
        for (int j = 0; j < 14; ++j) { e[j] = __expf(e[j] - mx); s += e[j]; }
        const float inv = 1.0f / s;
#pragma unroll
        for (int j = 0; j < 14; ++j) sAttn[i * 14 + j] = e[j] * inv;
    }
    __syncthreads();

    if (tid < 196) attn_out[(size_t)b * 196 + tid] = sAttn[tid];

    // h[i,d] = sum_j attn[i,j] * Wh[j,d]; ELU between layers; store bf16
    for (int idx = tid; idx < 14 * 768; idx += 256) {
        const int i = idx / 768;
        const int d = idx - i * 768;
        float s = 0.0f;
#pragma unroll
        for (int j = 0; j < 14; ++j) s += sAttn[i * 14 + j] * sWh[j * 768 + d];
        if (apply_elu && s < 0.0f) s = __expf(s) - 1.0f;
        x_out[base + idx] = f32_to_bf16_rne(s);
    }
}

// ---------------------------------------------------------------------------
// Orchestration. Workspace layout:
//   [0, 88MB)         X   : activations, bf16 (57344x768)
//   [88MB, 264MB)     WH  : Wh scratch, fp32  (57344x768)
//   [264MB, +5.9MB)   bf16 weights: W_emb | gat_W0 | gat_W1 | gat_W2 | W_out
// d_out layout (tuple): out (57344x768 f32) | attn0 | attn1 | attn2 (4096x196)
// ---------------------------------------------------------------------------
extern "C" void kernel_launch(void* const* d_in, const int* in_sizes, int n_in,
                              void* d_out, int out_size, void* d_ws, size_t ws_size,
                              hipStream_t stream) {
    const float* features = (const float*)d_in[0];
    const float* adj      = (const float*)d_in[1];
    const float* W_emb    = (const float*)d_in[2];
    const float* b_emb    = (const float*)d_in[3];
    const float* gat_W    = (const float*)d_in[4];
    const float* gat_a    = (const float*)d_in[5];
    const float* W_out    = (const float*)d_in[6];
    const float* b_out    = (const float*)d_in[7];
    float* out = (float*)d_out;

    const size_t MROWS = (size_t)4096 * 14;       // 57344
    const size_t XSZ   = MROWS * 768;             // activation elements
    const size_t WSZ   = (size_t)768 * 768;       // weight elements

    char* ws = (char*)d_ws;
    unsigned short* X   = (unsigned short*)ws;                 // bf16 activations
    float* WH           = (float*)(ws + XSZ * 2);              // fp32 Wh
    unsigned short* wb  = (unsigned short*)(ws + XSZ * 2 + XSZ * 4);

    // 1) weights -> bf16
    dim3 cgrid((unsigned)((WSZ + 255) / 256));
    f32_to_bf16_kernel<<<cgrid, 256, 0, stream>>>(W_emb, wb + 0 * WSZ, (int)WSZ);
    for (int l = 0; l < 3; ++l)
        f32_to_bf16_kernel<<<cgrid, 256, 0, stream>>>(gat_W + (size_t)l * WSZ,
                                                      wb + (size_t)(1 + l) * WSZ, (int)WSZ);
    f32_to_bf16_kernel<<<cgrid, 256, 0, stream>>>(W_out, wb + 4 * WSZ, (int)WSZ);

    dim3 ggrid(768 / 128, (unsigned)(MROWS / 128));   // (6, 448)

    // 2) x = features @ W_emb + b_emb   (fp32 A -> bf16 X)
    gemm_wmma_768<false, true><<<ggrid, 256, 0, stream>>>(features, wb + 0 * WSZ,
                                                          b_emb, X, 1);

    // 3) GAT layers: Wh = X @ W (bf16 A -> fp32 WH), fused attention writes X (bf16)
    float* attn_base = out + XSZ;
    for (int l = 0; l < 3; ++l) {
        gemm_wmma_768<true, false><<<ggrid, 256, 0, stream>>>(X, wb + (size_t)(1 + l) * WSZ,
                                                              nullptr, WH, 0);
        gat_attention<<<4096, 256, 0, stream>>>(WH, adj, gat_a + (size_t)l * 1536, X,
                                                attn_base + (size_t)l * 4096 * 196,
                                                (l < 2) ? 1 : 0);
    }

    // 4) out = x @ W_out + b_out   (bf16 A -> fp32 out)
    gemm_wmma_768<true, false><<<ggrid, 256, 0, stream>>>(X, wb + 4 * WSZ, b_out, out, 1);
}
